// MultiHeadSelfAttention_2705829396835
// MI455X (gfx1250) — matrix operand order repaired and checked by reading, not compiled, not run on hardware
//
#include <hip/hip_runtime.h>
#include <hip/hip_bf16.h>

typedef __attribute__((ext_vector_type(16))) _Float16 v16h;
typedef __attribute__((ext_vector_type(8)))  _Float16 v8h;
typedef __attribute__((ext_vector_type(4)))  _Float16 v4h;
typedef __attribute__((ext_vector_type(8)))  float    v8f;

// ---------------------------------------------------------------------------
// WMMA fragment helpers (CDNA5 gfx1250, wave32, v_wmma_f32_16x16x32_f16)
// ---------------------------------------------------------------------------

__device__ __forceinline__ v8f wmma_f16(v16h a, v16h b, v8f c) {
    // (neg_a, A, neg_b, B, c_mod, C, reuse_a, reuse_b)
    return __builtin_amdgcn_wmma_f32_16x16x32_f16(false, a, false, b,
                                                  (short)0, c, false, false);
}

// A-matrix 16x32 f16, row-major with leading dim `lda` (halves).
// Layout: lanes 0-15 row m=lane, elems 0-15 -> K=0..15;
//         lanes 16-31 row m=lane-16, K=16..31.
__device__ __forceinline__ v16h load_a_frag(const _Float16* base, int lda) {
    const int l  = threadIdx.x & 31;
    const int hi = l >> 4;
    const int m  = l & 15;
    const _Float16* p = base + (size_t)m * lda + 16 * hi;
    v8h x0 = *(const v8h*)(p);        // K = 16*hi + 0..7
    v8h x1 = *(const v8h*)(p + 8);    // K = 16*hi + 8..15
    return __builtin_shufflevector(x0, x1, 0,1,2,3,4,5,6,7,8,9,10,11,12,13,14,15);
}

// B-matrix 32x16 f16 where element (k, n) lives at base[n*ldn + k]
// ("K-contiguous" storage). Lanes 0-15: col n=lane, K=0..15; lanes 16-31: K=16..31.
__device__ __forceinline__ v16h load_b_frag_kcont(const _Float16* base, int ldn) {
    const int l  = threadIdx.x & 31;
    const int hi = l >> 4;
    const int n  = l & 15;
    const _Float16* p = base + (size_t)n * ldn + 16 * hi;
    v8h x0 = *(const v8h*)(p);        // K = 16*hi + 0..7
    v8h x1 = *(const v8h*)(p + 8);    // K = 16*hi + 8..15
    return __builtin_shufflevector(x0, x1, 0,1,2,3,4,5,6,7,8,9,10,11,12,13,14,15);
}

// ---------------------------------------------------------------------------
// Problem constants
// ---------------------------------------------------------------------------
#define E_DIM   1024
#define H_DIM   16
#define DH_DIM  64
#define S_DIM   2048
#define B_DIM   2
#define M_TOK   (B_DIM * S_DIM)   // 4096 tokens

// ---------------------------------------------------------------------------
// Conversion kernels
// ---------------------------------------------------------------------------
__global__ void cvt_f32_to_f16_kernel(const float* __restrict__ in,
                                      _Float16* __restrict__ out, int n) {
    int i = (blockIdx.x * blockDim.x + threadIdx.x) * 4;
    if (i < n) {
        float4 v = *(const float4*)(in + i);
        v4h h = { (_Float16)v.x, (_Float16)v.y, (_Float16)v.z, (_Float16)v.w };
        *(v4h*)(out + i) = h;
    }
}

// W [1024][1024] row-major (k, n)  ->  Wt [n][k] as f16
__global__ void transpose_cvt_kernel(const float* __restrict__ W,
                                     _Float16* __restrict__ outT) {
    int idx = blockIdx.x * blockDim.x + threadIdx.x;   // 1M threads
    int k = idx >> 10;
    int n = idx & 1023;
    outT[(size_t)n * E_DIM + k] = (_Float16)W[idx];
}

// ---------------------------------------------------------------------------
// Fused Q/K/V projection GEMM: [4096 x 1024] @ [1024 x 1024]
// Wave tile 64x64 (16 WMMAs per k-step, 1 b128 load per WMMA).
// blockIdx.z selects {Q, K, V}. Q is pre-scaled by 1/sqrt(Dh)=0.125.
// Q,K stored [bh][s][d] (f16);  V stored transposed [bh][d][s] (f16).
// ---------------------------------------------------------------------------
__global__ __launch_bounds__(128) void qkv_gemm_kernel(
    const _Float16* __restrict__ x16,
    const _Float16* __restrict__ wqT, const _Float16* __restrict__ wkT,
    const _Float16* __restrict__ wvT,
    const float* __restrict__ bq, const float* __restrict__ bk,
    const float* __restrict__ bv,
    _Float16* __restrict__ q16, _Float16* __restrict__ k16,
    _Float16* __restrict__ vT16)
{
    const int which = blockIdx.z;
    const _Float16* wT   = (which == 0) ? wqT : (which == 1) ? wkT : wvT;
    const float*    bias = (which == 0) ? bq  : (which == 1) ? bk  : bv;

    const int wave = threadIdx.x >> 5;
    const int n0 = blockIdx.x * 128 + (wave & 1) * 64;
    const int m0 = blockIdx.y * 128 + (wave >> 1) * 64;

    v8f acc[4][4] = {};
    for (int k0 = 0; k0 < E_DIM; k0 += 32) {
        v16h af[4], bf[4];
        #pragma unroll
        for (int i = 0; i < 4; ++i)
            af[i] = load_a_frag(x16 + (size_t)(m0 + 16 * i) * E_DIM + k0, E_DIM);
        #pragma unroll
        for (int j = 0; j < 4; ++j)
            bf[j] = load_b_frag_kcont(wT + (size_t)(n0 + 16 * j) * E_DIM + k0, E_DIM);
        #pragma unroll
        for (int i = 0; i < 4; ++i)
            #pragma unroll
            for (int j = 0; j < 4; ++j)
                acc[i][j] = wmma_f16(af[i], bf[j], acc[i][j]);
    }

    const int l  = threadIdx.x & 31;
    const int hi = l >> 4;
    const int nl = l & 15;
    const float scale = (which == 0) ? 0.125f : 1.0f;   // 1/sqrt(64) folded into Q

    for (int mi = 0; mi < 4; ++mi)
    for (int ni = 0; ni < 4; ++ni) {
        const int n  = n0 + ni * 16 + nl;
        const float bval = bias[n];
        const int h = n >> 6, d = n & 63;
        #pragma unroll
        for (int r = 0; r < 8; ++r) {
            const int m = m0 + mi * 16 + r + 8 * hi;     // token index
            const int bb = m >> 11, s = m & (S_DIM - 1);
            const int bh = bb * H_DIM + h;
            const float val = (acc[mi][ni][r] + bval) * scale;
            if (which == 2) {
                vT16[((size_t)bh * DH_DIM + d) * S_DIM + s] = (_Float16)val;
            } else {
                _Float16* dst = (which == 0) ? q16 : k16;
                dst[((size_t)bh * S_DIM + s) * DH_DIM + d] = (_Float16)val;
            }
        }
    }
}

// ---------------------------------------------------------------------------
// Flash attention: one block = 64 query rows of one (b,h); wave owns 16 rows.
// Online softmax over key blocks of 32; P staged via per-wave LDS tile.
// Writes attn output f16 in [b][s][h*64+d] layout (= [4096][1024]).
// ---------------------------------------------------------------------------
__global__ __launch_bounds__(128) void attn_kernel(
    const _Float16* __restrict__ q16, const _Float16* __restrict__ k16,
    const _Float16* __restrict__ vT16, _Float16* __restrict__ attn16)
{
    __shared__ __align__(16) float    sS[4][16][32];
    __shared__ __align__(16) _Float16 sP[4][16][32];
    __shared__ __align__(16) float    sA[4][16];

    const int wave = threadIdx.x >> 5;
    const int l    = threadIdx.x & 31;
    const int hi   = l >> 4;
    const int nl   = l & 15;
    const int bh   = blockIdx.y;                 // 0..31
    const int q0   = blockIdx.x * 64 + wave * 16;

    const _Float16* qbase = q16 + ((size_t)bh * S_DIM + q0) * DH_DIM;
    const v16h aq0 = load_a_frag(qbase, DH_DIM);        // d = 0..31
    const v16h aq1 = load_a_frag(qbase + 32, DH_DIM);   // d = 32..63

    v8f o[4] = {};                    // output accum, d-tiles {0,16,32,48}
    float m_run = -1e30f, l_run = 0.0f;

    for (int j0 = 0; j0 < S_DIM; j0 += 32) {
        // ---- scores = Qs . K^T for 32 keys (two 16-col tiles) ----
        #pragma unroll
        for (int t = 0; t < 2; ++t) {
            const _Float16* kb = k16 + ((size_t)bh * S_DIM + j0 + t * 16) * DH_DIM;
            v16h bk0 = load_b_frag_kcont(kb, DH_DIM);        // d 0..31
            v16h bk1 = load_b_frag_kcont(kb + 32, DH_DIM);   // d 32..63
            v8f s = {};
            s = wmma_f16(aq0, bk0, s);
            s = wmma_f16(aq1, bk1, s);
            #pragma unroll
            for (int r = 0; r < 8; ++r)
                sS[wave][r + 8 * hi][t * 16 + nl] = s[r];
        }
        __syncthreads();

        // ---- online softmax stats: lanes 0-15 each own one row ----
        if (hi == 0) {
            const int r = nl;
            float mx = -1e30f;
            #pragma unroll 8
            for (int c = 0; c < 32; ++c) mx = fmaxf(mx, sS[wave][r][c]);
            const float nm = fmaxf(m_run, mx);
            const float alpha = __expf(m_run - nm);
            float sum = 0.0f;
            #pragma unroll 8
            for (int c = 0; c < 32; ++c) {
                const float p = __expf(sS[wave][r][c] - nm);
                sum += p;
                sP[wave][r][c] = (_Float16)p;
            }
            l_run = l_run * alpha + sum;
            m_run = nm;
            sA[wave][r] = alpha;
        }
        __syncthreads();

        // ---- rescale accumulators, then O += P @ V ----
        #pragma unroll
        for (int t = 0; t < 4; ++t)
            #pragma unroll
            for (int r = 0; r < 8; ++r)
                o[t][r] *= sA[wave][r + 8 * hi];

        const v16h ap = load_a_frag(&sP[wave][0][0], 32);    // 16x32 P tile
        #pragma unroll
        for (int t = 0; t < 4; ++t) {
            const _Float16* vb = vT16 + ((size_t)bh * DH_DIM + t * 16) * S_DIM + j0;
            v16h bv = load_b_frag_kcont(vb, S_DIM);          // keys contiguous
            o[t] = wmma_f16(ap, bv, o[t]);
        }
    }

    // ---- normalize by softmax denominator and store ----
    if (hi == 0) sA[wave][nl] = 1.0f / l_run;
    __syncthreads();

    const int bb = bh >> 4, h = bh & 15;
    #pragma unroll
    for (int t = 0; t < 4; ++t) {
        const int d = t * 16 + nl;
        #pragma unroll
        for (int r = 0; r < 8; ++r) {
            const int m = r + 8 * hi;
            const float val = o[t][r] * sA[wave][m];
            const int srow = q0 + m;
            attn16[((size_t)bb * S_DIM + srow) * E_DIM + h * DH_DIM + d] = (_Float16)val;
        }
    }
}

// ---------------------------------------------------------------------------
// Output projection: out = attn @ Wo + bo   (fp32 output)
// Wave tile 64x64 (16 WMMAs per k-step).
// ---------------------------------------------------------------------------
__global__ __launch_bounds__(128) void out_gemm_kernel(
    const _Float16* __restrict__ a16, const _Float16* __restrict__ woT,
    const float* __restrict__ bo, float* __restrict__ out)
{
    const int wave = threadIdx.x >> 5;
    const int n0 = blockIdx.x * 128 + (wave & 1) * 64;
    const int m0 = blockIdx.y * 128 + (wave >> 1) * 64;

    v8f acc[4][4] = {};
    for (int k0 = 0; k0 < E_DIM; k0 += 32) {
        v16h af[4], bf[4];
        #pragma unroll
        for (int i = 0; i < 4; ++i)
            af[i] = load_a_frag(a16 + (size_t)(m0 + 16 * i) * E_DIM + k0, E_DIM);
        #pragma unroll
        for (int j = 0; j < 4; ++j)
            bf[j] = load_b_frag_kcont(woT + (size_t)(n0 + 16 * j) * E_DIM + k0, E_DIM);
        #pragma unroll
        for (int i = 0; i < 4; ++i)
            #pragma unroll
            for (int j = 0; j < 4; ++j)
                acc[i][j] = wmma_f16(af[i], bf[j], acc[i][j]);
    }

    const int l  = threadIdx.x & 31;
    const int hi = l >> 4;
    const int nl = l & 15;
    for (int mi = 0; mi < 4; ++mi)
    for (int ni = 0; ni < 4; ++ni) {
        const int n = n0 + ni * 16 + nl;
        const float bval = bo[n];
        #pragma unroll
        for (int r = 0; r < 8; ++r) {
            const int m = m0 + mi * 16 + r + 8 * hi;
            out[(size_t)m * E_DIM + n] = acc[mi][ni][r] + bval;
        }
    }
}

// ---------------------------------------------------------------------------
// Host launcher
// ---------------------------------------------------------------------------
extern "C" void kernel_launch(void* const* d_in, const int* in_sizes, int n_in,
                              void* d_out, int out_size, void* d_ws, size_t ws_size,
                              hipStream_t stream) {
    const float* x  = (const float*)d_in[0];
    const float* Wq = (const float*)d_in[1];
    const float* bq = (const float*)d_in[2];
    const float* Wk = (const float*)d_in[3];
    const float* bk = (const float*)d_in[4];
    const float* Wv = (const float*)d_in[5];
    const float* bv = (const float*)d_in[6];
    const float* Wo = (const float*)d_in[7];
    const float* bo = (const float*)d_in[8];
    float* out = (float*)d_out;

    char* ws = (char*)d_ws;
    const size_t MB = 1024 * 1024;
    _Float16* x16  = (_Float16*)(ws);             //  8 MB  [4096][1024]
    _Float16* wqT  = (_Float16*)(ws +  8 * MB);   //  2 MB  [n][k]
    _Float16* wkT  = (_Float16*)(ws + 10 * MB);   //  2 MB
    _Float16* wvT  = (_Float16*)(ws + 12 * MB);   //  2 MB
    _Float16* woT  = (_Float16*)(ws + 14 * MB);   //  2 MB
    _Float16* q16  = (_Float16*)(ws + 16 * MB);   //  8 MB  [bh][s][d] (pre-scaled)
    _Float16* k16  = (_Float16*)(ws + 24 * MB);   //  8 MB  [bh][s][d]
    _Float16* vT16 = (_Float16*)(ws + 32 * MB);   //  8 MB  [bh][d][s]
    _Float16* a16  = (_Float16*)(ws + 40 * MB);   //  8 MB  [4096][1024]

    // 1) convert inputs to f16 (weights transposed for K-contiguous B frags)
    cvt_f32_to_f16_kernel<<<4096, 256, 0, stream>>>(x, x16, M_TOK * E_DIM);
    transpose_cvt_kernel<<<4096, 256, 0, stream>>>(Wq, wqT);
    transpose_cvt_kernel<<<4096, 256, 0, stream>>>(Wk, wkT);
    transpose_cvt_kernel<<<4096, 256, 0, stream>>>(Wv, wvT);
    transpose_cvt_kernel<<<4096, 256, 0, stream>>>(Wo, woT);

    // 2) fused Q/K/V projections (z selects which)
    qkv_gemm_kernel<<<dim3(E_DIM / 128, M_TOK / 128, 3), 128, 0, stream>>>(
        x16, wqT, wkT, wvT, bq, bk, bv, q16, k16, vT16);

    // 3) flash attention
    attn_kernel<<<dim3(S_DIM / 64, B_DIM * H_DIM), 128, 0, stream>>>(
        q16, k16, vT16, a16);

    // 4) output projection
    out_gemm_kernel<<<dim3(E_DIM / 128, M_TOK / 128), 128, 0, stream>>>(
        a16, woT, bo, out);
}
